// MCFModel_30915174596960
// MI455X (gfx1250) — matrix operand order repaired and checked by reading, not compiled, not run on hardware
//
#include <hip/hip_runtime.h>

#define BB 8
#define VV 1024
#define EE 32
#define FF 64
#define HH 4
#define HIDD 64
#define NLAYERS 2
#define NFLOW 10

typedef __attribute__((ext_vector_type(16))) _Float16 v16h;
typedef __attribute__((ext_vector_type(8)))  _Float16 v8h;
typedef __attribute__((ext_vector_type(8)))  float    v8f;

// ---------------- WMMA fragment helpers (CDNA5 16x16x32 f16 layouts) -------
// A fragment (16x32 f16): lane m=lane&15,h=lane>>4; elem e -> K = 16*(e>>3)+8h+(e&7)
// Two contiguous runs of 8 halves -> two b128 loads. Requires ld % 8 == 0.
__device__ __forceinline__ v16h load_a_frag(const _Float16* A, int ld) {
  const int lane = threadIdx.x & 31;
  const int m = lane & 15, hh = lane >> 4;
  const _Float16* p = A + m * ld + 8 * hh;
  v8h lo = *(const v8h*)p;
  v8h hi = *(const v8h*)(p + 16);
  v16h r;
#pragma unroll
  for (int e = 0; e < 8; ++e) { r[e] = lo[e]; r[e + 8] = hi[e]; }
  return r;
}
// B fragment (32x16 f16) from an N-major (transposed) matrix Bt[n][k], ld = K stride.
// lane n=lane&15,h=lane>>4; elem e -> K = 16*(e>>3)+8h+(e&7) -> two runs of 8 halves.
__device__ __forceinline__ v16h load_bt_frag(const _Float16* Bt, int ld) {
  const int lane = threadIdx.x & 31;
  const int n = lane & 15, hh = lane >> 4;
  const _Float16* p = Bt + n * ld + 8 * hh;
  v8h lo = *(const v8h*)p;
  v8h hi = *(const v8h*)(p + 16);
  v16h r;
#pragma unroll
  for (int e = 0; e < 8; ++e) { r[e] = lo[e]; r[e + 8] = hi[e]; }
  return r;
}

// ---------------- weight prep: f32 -> f16 transposed copies ----------------
__global__ void k_prep(const float* __restrict__ gat_w, const float* __restrict__ dec_w2,
                       _Float16* __restrict__ gat_wT, _Float16* __restrict__ dec_w2T) {
  int i = blockIdx.x * blockDim.x + threadIdx.x;
  if (i < HH * FF * FF) {          // gat_wT[h][g][f] = gat_w[h][f][g]
    int h = i / (FF * FF); int r = i % (FF * FF); int g = r / FF; int f = r % FF;
    gat_wT[i] = (_Float16)gat_w[(h * FF + f) * FF + g];
  }
  if (i < VV * HIDD) {             // dec_w2T[n][k] = dec_w2[k][n]
    int n = i / HIDD; int k = i % HIDD;
    dec_w2T[i] = (_Float16)dec_w2[k * VV + n];
  }
}

// ---------------- encoder MLP: per (b,v) row, 64 threads -------------------
__global__ void k_encoder(const float* __restrict__ dem, const float* __restrict__ emb,
                          const float* __restrict__ w1, const float* __restrict__ b1,
                          const float* __restrict__ w2, const float* __restrict__ b2,
                          float* __restrict__ enc, _Float16* __restrict__ enc16) {
  __shared__ float xS[EE + 1];
  __shared__ float hS[HIDD];
  int row = blockIdx.x;                  // b*V + v
  int v = row & (VV - 1);
  int t = threadIdx.x;
  if (t < EE) xS[t] = emb[v * EE + t];
  if (t == 0) xS[EE] = dem[row];
  __syncthreads();
  float h = b1[t];
#pragma unroll
  for (int i = 0; i < EE + 1; ++i) h += xS[i] * w1[i * HIDD + t];
  h = fmaxf(h, 0.f);
  hS[t] = h;
  __syncthreads();
  float e = b2[t];
#pragma unroll
  for (int j = 0; j < HIDD; ++j) e += hS[j] * w2[j * FF + t];
  e = fmaxf(e, 0.f);
  enc[(size_t)row * FF + t] = e;
  enc16[(size_t)row * FF + t] = (_Float16)e;
}

// ---------------- t = enc @ gat_w[h], output transposed tT[b,h][g][v] (f16) -
__global__ void k_tgemm(const _Float16* __restrict__ enc16,
                        const _Float16* __restrict__ gat_wT,
                        _Float16* __restrict__ tT) {
  int tile = blockIdx.x & 63;            // v-tile (V/16 = 64)
  int bh = blockIdx.x >> 6;              // b*H + h
  int b = bh >> 2, h = bh & 3;
  int v0 = tile << 4;
  const _Float16* A  = enc16 + (size_t)(b * VV + v0) * FF;
  const _Float16* Bt = gat_wT + (size_t)h * FF * FF;
  const int lane = threadIdx.x & 31;
  const int n = lane & 15, hh = lane >> 4;
  v8f c[4];
#pragma unroll
  for (int i = 0; i < 4; ++i)
#pragma unroll
    for (int r = 0; r < 8; ++r) c[i][r] = 0.f;
#pragma unroll
  for (int k0 = 0; k0 < FF; k0 += 32) {
    v16h a = load_a_frag(A + k0, FF);
#pragma unroll
    for (int i = 0; i < 4; ++i) {
      v16h bf = load_bt_frag(Bt + (i * 16) * FF + k0, FF);
      c[i] = __builtin_amdgcn_wmma_f32_16x16x32_f16(false, a, false, bf,
                                                    (short)0, c[i], false, false);
    }
  }
  _Float16* out = tT + (size_t)bh * FF * VV;
#pragma unroll
  for (int i = 0; i < 4; ++i) {
    v8h o;
#pragma unroll
    for (int r = 0; r < 8; ++r) o[r] = (_Float16)c[i][r];
    *(v8h*)(out + (size_t)(i * 16 + n) * VV + v0 + 8 * hh) = o;    // rows m=8hh+r contiguous in v
  }
}

// ---------------- s_self / s_nbr dots --------------------------------------
__global__ void k_sdots(const _Float16* __restrict__ tT,
                        const float* __restrict__ a1, const float* __restrict__ a2,
                        float* __restrict__ sself, float* __restrict__ snbr) {
  int bh = blockIdx.x >> 2;
  int v = ((blockIdx.x & 3) << 8) + threadIdx.x;
  int h = bh & 3;
  const _Float16* tp = tT + (size_t)bh * FF * VV + v;
  float s1 = 0.f, s2 = 0.f;
#pragma unroll
  for (int g = 0; g < FF; ++g) {
    float tv = (float)tp[(size_t)g * VV];
    s1 += tv * a1[h * FF + g];
    s2 += tv * a2[h * FF + g];
  }
  sself[bh * VV + v] = s1;
  snbr[bh * VV + v] = s2;
}

// ---------------- softmax stats (rank-1 logits -> no t needed) -------------
__global__ void k_stats(const float* __restrict__ sself, const float* __restrict__ snbr,
                        const float* __restrict__ adj,
                        float* __restrict__ mrow, float* __restrict__ den) {
  __shared__ float red[128];
  int bhv = blockIdx.x;
  int v = bhv & (VV - 1);
  int bh = bhv >> 10;
  int t = threadIdx.x;
  float ss = sself[bhv];
  const float* adjr = adj + (size_t)v * VV;
  const float* snr = snbr + bh * VV;
  float m = -3.0e38f;
  for (int w = t; w < VV; w += 128) {
    if (adjr[w] > 0.f) {
      float l = ss + snr[w];
      l = l > 0.f ? l : 0.2f * l;
      m = fmaxf(m, l);
    }
  }
  red[t] = m; __syncthreads();
  for (int s = 64; s > 0; s >>= 1) { if (t < s) red[t] = fmaxf(red[t], red[t + s]); __syncthreads(); }
  m = red[0]; __syncthreads();
  float d = 0.f;
  for (int w = t; w < VV; w += 128) {
    if (adjr[w] > 0.f) {
      float l = ss + snr[w];
      l = l > 0.f ? l : 0.2f * l;
      d += __expf(l - m);
    }
  }
  red[t] = d; __syncthreads();
  for (int s = 64; s > 0; s >>= 1) { if (t < s) red[t] += red[t + s]; __syncthreads(); }
  if (t == 0) { mrow[bhv] = m; den[bhv] = red[0]; }
}

// ---------------- flash-style heads = softmax(P) @ t (WMMA) ----------------
__global__ void k_heads(const float* __restrict__ sself, const float* __restrict__ snbr,
                        const float* __restrict__ adj,
                        const float* __restrict__ mrow, const float* __restrict__ den,
                        const _Float16* __restrict__ tT, float* __restrict__ heads) {
  int tile = blockIdx.x & 63;
  int bh = blockIdx.x >> 6;
  int v0 = tile << 4;
  const int lane = threadIdx.x & 31;
  const int m = lane & 15, hh = lane >> 4;
  const int v = v0 + m;
  float ss = sself[bh * VV + v];
  float mr = mrow[bh * VV + v];
  const float* snr  = snbr + bh * VV;
  const float* adjr = adj + (size_t)v * VV;
  const _Float16* tTb = tT + (size_t)bh * FF * VV;
  v8f c[4];
#pragma unroll
  for (int i = 0; i < 4; ++i)
#pragma unroll
    for (int r = 0; r < 8; ++r) c[i][r] = 0.f;
  for (int w0 = 0; w0 < VV; w0 += 32) {
    v16h a;
#pragma unroll
    for (int half = 0; half < 2; ++half) {       // regenerate P tile in registers
      int wb = w0 + 16 * half + 8 * hh;
      const float4* snp = (const float4*)(snr + wb);
      const float4* adp = (const float4*)(adjr + wb);
      float4 sn0 = snp[0], sn1 = snp[1];
      float4 ad0 = adp[0], ad1 = adp[1];
      float snv[8] = {sn0.x, sn0.y, sn0.z, sn0.w, sn1.x, sn1.y, sn1.z, sn1.w};
      float adv[8] = {ad0.x, ad0.y, ad0.z, ad0.w, ad1.x, ad1.y, ad1.z, ad1.w};
#pragma unroll
      for (int j = 0; j < 8; ++j) {
        float l = ss + snv[j];
        l = l > 0.f ? l : 0.2f * l;
        float p = adv[j] > 0.f ? __expf(l - mr) : 0.f;
        a[half * 8 + j] = (_Float16)p;
      }
    }
#pragma unroll
    for (int i = 0; i < 4; ++i) {
      v16h bf = load_bt_frag(tTb + (size_t)(i * 16) * VV + w0, VV);
      c[i] = __builtin_amdgcn_wmma_f32_16x16x32_f16(false, a, false, bf,
                                                    (short)0, c[i], false, false);
    }
  }
  float invd[8];
#pragma unroll
  for (int r = 0; r < 8; ++r) invd[r] = 1.f / den[bh * VV + v0 + 8 * hh + r];
  float* out = heads + (size_t)bh * VV * FF;
#pragma unroll
  for (int i = 0; i < 4; ++i)
#pragma unroll
    for (int r = 0; r < 8; ++r)
      out[(size_t)(v0 + 8 * hh + r) * FF + i * 16 + (lane & 15)] = c[i][r] * invd[r];
}

// ---------------- head mean + GRU-style gate -------------------------------
__global__ void k_gate(const float* __restrict__ heads, const float* __restrict__ enc_in,
                       const float* __restrict__ gate_w, const float* __restrict__ gate_u,
                       const float* __restrict__ gate_b,
                       float* __restrict__ enc_out, _Float16* __restrict__ enc_out16) {
  __shared__ float nxtS[FF], encS[FF];
  int row = blockIdx.x;                   // b*V + v
  int b = row >> 10, v = row & (VV - 1);
  int g = threadIdx.x;
  float acc = 0.f;
#pragma unroll
  for (int h = 0; h < HH; ++h) acc += heads[((size_t)(b * HH + h) * VV + v) * FF + g];
  float nxt = fmaxf(acc * 0.25f, 0.f);
  float e_in = enc_in[(size_t)row * FF + g];
  nxtS[g] = nxt; encS[g] = e_in;
  __syncthreads();
  float zin = gate_b[g];
#pragma unroll
  for (int f = 0; f < FF; ++f) zin += nxtS[f] * gate_w[f * FF + g] + encS[f] * gate_u[f * FF + g];
  float z = 1.f / (1.f + __expf(-zin));
  float o = z * nxt + (1.f - z) * e_in;
  enc_out[(size_t)row * FF + g] = o;
  enc_out16[(size_t)row * FF + g] = (_Float16)o;
}

// ---------------- decoder hidden (linear) ----------------------------------
__global__ void k_d1(const float* __restrict__ enc, const float* __restrict__ w1,
                     const float* __restrict__ b1, _Float16* __restrict__ d116) {
  __shared__ float encS[FF];
  int row = blockIdx.x; int g = threadIdx.x;
  encS[g] = enc[(size_t)row * FF + g];
  __syncthreads();
  float d = b1[g];
#pragma unroll
  for (int f = 0; f < FF; ++f) d += encS[f] * w1[f * HIDD + g];
  d116[(size_t)row * HIDD + g] = (_Float16)d;
}

// ---------------- pred = square(d1 @ dec_w2 + b2) (WMMA) -------------------
__global__ void k_pred(const _Float16* __restrict__ d116, const _Float16* __restrict__ w2T,
                       const float* __restrict__ b2, float* __restrict__ pred) {
  int b   = blockIdx.x >> 10;
  int rem = blockIdx.x & 1023;
  int vt = rem >> 4, nb = rem & 15;
  int v0 = vt << 4, n0 = nb << 6;
  const _Float16* A = d116 + (size_t)(b * VV + v0) * HIDD;
  const int lane = threadIdx.x & 31;
  const int n = lane & 15, hh = lane >> 4;
  v8f c[4];
#pragma unroll
  for (int i = 0; i < 4; ++i)
#pragma unroll
    for (int r = 0; r < 8; ++r) c[i][r] = 0.f;
#pragma unroll
  for (int k0 = 0; k0 < HIDD; k0 += 32) {
    v16h a = load_a_frag(A + k0, HIDD);
#pragma unroll
    for (int i = 0; i < 4; ++i) {
      v16h bf = load_bt_frag(w2T + (size_t)(n0 + i * 16) * HIDD + k0, HIDD);
      c[i] = __builtin_amdgcn_wmma_f32_16x16x32_f16(false, a, false, bf,
                                                    (short)0, c[i], false, false);
    }
  }
  float* out = pred + (size_t)b * VV * VV;
#pragma unroll
  for (int i = 0; i < 4; ++i) {
    float bias = b2[n0 + i * 16 + n];
#pragma unroll
    for (int r = 0; r < 8; ++r) {
      float pv = c[i][r] + bias;
      out[(size_t)(v0 + 8 * hh + r) * VV + n0 + i * 16 + n] = pv * pv;
    }
  }
}

// ---------------- masked row softmax over pred (in-place) + rowsq ----------
__global__ void k_fwsm(float* __restrict__ pred, const float* __restrict__ adj,
                       float* __restrict__ rowsq) {
  __shared__ float red[256];
  int row = blockIdx.x;                   // b*V + v
  int v = row & (VV - 1);
  int t = threadIdx.x;
  float* pr = pred + (size_t)row * VV;
  const float* adjr = adj + (size_t)v * VV;
  float m = -3.0e38f;
  for (int w = t; w < VV; w += 256) if (adjr[w] > 0.f) m = fmaxf(m, pr[w]);
  red[t] = m; __syncthreads();
  for (int s = 128; s > 0; s >>= 1) { if (t < s) red[t] = fmaxf(red[t], red[t + s]); __syncthreads(); }
  m = red[0]; __syncthreads();
  float d = 0.f;
  for (int w = t; w < VV; w += 256) if (adjr[w] > 0.f) d += __expf(pr[w] - m);
  red[t] = d; __syncthreads();
  for (int s = 128; s > 0; s >>= 1) { if (t < s) red[t] += red[t + s]; __syncthreads(); }
  float inv = 1.f / red[0]; __syncthreads();
  float sq = 0.f;
  for (int w = t; w < VV; w += 256) {
    float fwv = adjr[w] > 0.f ? __expf(pr[w] - m) * inv : 0.f;
    pr[w] = fwv;
    sq += fwv * fwv;
  }
  red[t] = sq; __syncthreads();
  for (int s = 128; s > 0; s >>= 1) { if (t < s) red[t] += red[t + s]; __syncthreads(); }
  if (t == 0) rowsq[row] = red[0];
}

// ---------------- flow solver: 10 iterations in one workgroup per batch ----
// s_{k+1}[v] = relu( sum_u fw[u][v]*s_k[u] - dem[v] );  fw slice is L2-resident
__global__ void k_flow(const float* __restrict__ fw, const float* __restrict__ dem,
                       const float* __restrict__ rowsq, float* __restrict__ fcost) {
  __shared__ float s[VV];
  __shared__ float red[VV];
  int b = blockIdx.x, v = threadIdx.x;
  float d = dem[b * VV + v];
  s[v] = fmaxf(-d, 0.f);
  __syncthreads();
  const float* fwb = fw + (size_t)b * VV * VV;
  for (int it = 0; it < NFLOW - 1; ++it) {
    float acc = 0.f;
    for (int u = 0; u < VV; ++u) acc += fwb[(size_t)u * VV + v] * s[u];
    float sn = fmaxf(acc - d, 0.f);
    __syncthreads();
    s[v] = sn;
    __syncthreads();
  }
  red[v] = s[v] * s[v] * rowsq[b * VV + v];   // flow_cost = sum_v rowsq*s^2
  __syncthreads();
  for (int st = 512; st > 0; st >>= 1) { if (v < st) red[v] += red[v + st]; __syncthreads(); }
  if (v == 0) fcost[b] = red[0];
}

// ---------------- dual variables -------------------------------------------
__global__ void k_dv(const float* __restrict__ enc, const float* __restrict__ w1,
                     const float* __restrict__ b1, const float* __restrict__ w2,
                     const float* __restrict__ b2, float* __restrict__ dv) {
  __shared__ float encS[FF];
  __shared__ float red[HIDD];
  int row = blockIdx.x; int j = threadIdx.x;
  encS[j] = enc[(size_t)row * FF + j];
  __syncthreads();
  float h = b1[j];
#pragma unroll
  for (int f = 0; f < FF; ++f) h += encS[f] * w1[f * HIDD + j];
  red[j] = h * w2[j];
  __syncthreads();
  for (int s = 32; s > 0; s >>= 1) { if (j < s) red[j] += red[j + s]; __syncthreads(); }
  if (j == 0) dv[row] = red[0] + b2[0];
}

__global__ void k_zero(float* __restrict__ dfc) { if (threadIdx.x < BB) dfc[threadIdx.x] = 0.f; }

// dual_flow_cost = -relu(dd)^2/4 on adjacent edges (analytic simplification)
__global__ void k_dualpair(const float* __restrict__ dv, const float* __restrict__ adj,
                           float* __restrict__ dfc) {
  __shared__ float red[256];
  int row = blockIdx.x; int b = row >> 10, v = row & (VV - 1);
  int t = threadIdx.x;
  float dvv = dv[row];
  const float* adjr = adj + (size_t)v * VV;
  const float* dvb = dv + b * VV;
  float acc = 0.f;
  for (int w = t; w < VV; w += 256) {
    float dd = dvv - dvb[w];
    if (adjr[w] > 0.f && dd > 0.f) acc -= 0.25f * dd * dd;
  }
  red[t] = acc; __syncthreads();
  for (int s = 128; s > 0; s >>= 1) { if (t < s) red[t] += red[t + s]; __syncthreads(); }
  if (t == 0) atomicAdd(&dfc[b], red[0]);
}

__global__ void k_final(const float* __restrict__ fcost, const float* __restrict__ dfc,
                        const float* __restrict__ dv, const float* __restrict__ dem,
                        float* __restrict__ out) {
  __shared__ float red[256];
  int t = threadIdx.x;
  float loss = 0.f;
  for (int b = 0; b < BB; ++b) {
    float dd = 0.f;
    for (int v = t; v < VV; v += 256) dd += dv[b * VV + v] * dem[b * VV + v];
    red[t] = dd; __syncthreads();
    for (int s = 128; s > 0; s >>= 1) { if (t < s) red[t] += red[t + s]; __syncthreads(); }
    if (t == 0) {
      float dual_cost = dfc[b] - red[0];
      loss += fcost[b] - dual_cost;
    }
    __syncthreads();
  }
  if (t == 0) out[0] = loss / (float)BB;
}

// ---------------------------------------------------------------------------
extern "C" void kernel_launch(void* const* d_in, const int* in_sizes, int n_in,
                              void* d_out, int out_size, void* d_ws, size_t ws_size,
                              hipStream_t stream) {
  (void)in_sizes; (void)n_in; (void)out_size; (void)ws_size;
  const float* demands = (const float*)d_in[0];
  const float* emb     = (const float*)d_in[1];
  const float* adj     = (const float*)d_in[2];
  const float* enc_w1  = (const float*)d_in[3];
  const float* enc_b1  = (const float*)d_in[4];
  const float* enc_w2  = (const float*)d_in[5];
  const float* enc_b2  = (const float*)d_in[6];
  const float* gat_w   = (const float*)d_in[7];
  const float* gat_a1  = (const float*)d_in[8];
  const float* gat_a2  = (const float*)d_in[9];
  const float* gate_w  = (const float*)d_in[10];
  const float* gate_u  = (const float*)d_in[11];
  const float* gate_b  = (const float*)d_in[12];
  const float* dec_w1  = (const float*)d_in[13];
  const float* dec_b1  = (const float*)d_in[14];
  const float* dec_w2  = (const float*)d_in[15];
  const float* dec_b2  = (const float*)d_in[16];
  const float* dual_w1 = (const float*)d_in[17];
  const float* dual_b1 = (const float*)d_in[18];
  const float* dual_w2 = (const float*)d_in[19];
  const float* dual_b2 = (const float*)d_in[20];

  char* ws = (char*)d_ws;
  size_t off = 0;
  auto alloc = [&](size_t bytes) -> void* {
    void* p = ws + off;
    off = (off + bytes + 255) & ~(size_t)255;
    return p;
  };
  float*    encA    = (float*)   alloc((size_t)BB * VV * FF * 4);
  float*    encB    = (float*)   alloc((size_t)BB * VV * FF * 4);
  _Float16* encA16  = (_Float16*)alloc((size_t)BB * VV * FF * 2);
  _Float16* encB16  = (_Float16*)alloc((size_t)BB * VV * FF * 2);
  _Float16* tT      = (_Float16*)alloc((size_t)BB * HH * FF * VV * 2);
  float*    sself   = (float*)   alloc((size_t)BB * HH * VV * 4);
  float*    snbr    = (float*)   alloc((size_t)BB * HH * VV * 4);
  float*    mrowb   = (float*)   alloc((size_t)BB * HH * VV * 4);
  float*    denb    = (float*)   alloc((size_t)BB * HH * VV * 4);
  float*    headsb  = (float*)   alloc((size_t)BB * HH * VV * FF * 4);
  _Float16* d116    = (_Float16*)alloc((size_t)BB * VV * HIDD * 2);
  float*    predb   = (float*)   alloc((size_t)BB * VV * VV * 4);   // becomes fw in-place
  float*    rowsqb  = (float*)   alloc((size_t)BB * VV * 4);
  float*    dvb     = (float*)   alloc((size_t)BB * VV * 4);
  float*    fcostb  = (float*)   alloc(64);
  float*    dfcb    = (float*)   alloc(64);
  _Float16* gatwT   = (_Float16*)alloc((size_t)HH * FF * FF * 2);
  _Float16* decw2T  = (_Float16*)alloc((size_t)VV * HIDD * 2);

  k_prep<<<(VV * HIDD + 255) / 256, 256, 0, stream>>>(gat_w, dec_w2, gatwT, decw2T);
  k_encoder<<<BB * VV, 64, 0, stream>>>(demands, emb, enc_w1, enc_b1, enc_w2, enc_b2,
                                        encA, encA16);
  float* encCur = encA;  _Float16* encCur16 = encA16;
  float* encNxt = encB;  _Float16* encNxt16 = encB16;
  for (int layer = 0; layer < NLAYERS; ++layer) {
    k_tgemm<<<BB * HH * (VV / 16), 32, 0, stream>>>(encCur16, gatwT, tT);
    k_sdots<<<BB * HH * (VV / 256), 256, 0, stream>>>(tT, gat_a1, gat_a2, sself, snbr);
    k_stats<<<BB * HH * VV, 128, 0, stream>>>(sself, snbr, adj, mrowb, denb);
    k_heads<<<BB * HH * (VV / 16), 32, 0, stream>>>(sself, snbr, adj, mrowb, denb, tT, headsb);
    k_gate<<<BB * VV, 64, 0, stream>>>(headsb, encCur, gate_w, gate_u, gate_b,
                                       encNxt, encNxt16);
    float* tf = encCur; encCur = encNxt; encNxt = tf;
    _Float16* th = encCur16; encCur16 = encNxt16; encNxt16 = th;
  }
  k_d1<<<BB * VV, 64, 0, stream>>>(encCur, dec_w1, dec_b1, d116);
  k_pred<<<BB * (VV / 16) * (VV / 64), 32, 0, stream>>>(d116, decw2T, dec_b2, predb);
  k_fwsm<<<BB * VV, 256, 0, stream>>>(predb, adj, rowsqb);
  k_flow<<<BB, VV, 0, stream>>>(predb, demands, rowsqb, fcostb);
  k_dv<<<BB * VV, 64, 0, stream>>>(encCur, dual_w1, dual_b1, dual_w2, dual_b2, dvb);
  k_zero<<<1, 32, 0, stream>>>(dfcb);
  k_dualpair<<<BB * VV, 256, 0, stream>>>(dvb, adj, dfcb);
  k_final<<<1, 256, 0, stream>>>(fcostb, dfcb, dvb, demands, (float*)d_out);
}
